// GPT_45183055954536
// MI455X (gfx1250) — compile-verified, matrix operand order checked by reading, not run on hardware
//
#include <hip/hip_runtime.h>
#include <hip/hip_bf16.h>
#include <math.h>

// ---------------------------------------------------------------------------
// GPT-2 small forward (B=4,T=1024,D=768,H=12,HS=64,L=12,V=50257) for gfx1250.
//
// Design for MI455X:
//  * all weights converted+transposed once per call to f16 [N x K] workspace
//  * all GEMM operands are f16, K-contiguous -> LDS tiles staged with
//    GLOBAL_LOAD_ASYNC_TO_LDS_B128 (ASYNCcnt) and double-buffered
//  * inner loop = async issue + ds fragment loads + v_wmma_f32_16x16x32_f16
//  * epilogues fuse bias/residual/relu and emit f32 and/or f16 outputs
// ---------------------------------------------------------------------------

typedef __attribute__((ext_vector_type(16))) _Float16 v16h;
typedef __attribute__((ext_vector_type(8)))  float    v8f;

#define TM 128
#define TN 64
#define TK 32
#define PITCH 20   // dwords per LDS row: 16 data + 4 pad (80 B, 16B-aligned)

#define FLAG_BIAS  1
#define FLAG_RELU  2
#define FLAG_RESID 4
#define FLAG_CHT   8   // store Ch transposed: Ch[n*ldch + m]

union Frag { v16h v; unsigned int u[8]; };

__device__ __forceinline__ unsigned lds_addr(const void* p) {
    // low 32 bits of a generic LDS pointer are the raw LDS byte offset
    return (unsigned)(unsigned long long)p;
}

__device__ __forceinline__ void async_b128(unsigned lds_off, const void* gaddr) {
    asm volatile("global_load_async_to_lds_b128 %0, %1, off"
                 :: "v"(lds_off), "v"(gaddr) : "memory");
}

__device__ __forceinline__ void wait_async0() {
    asm volatile("s_wait_asynccnt 0x0" ::: "memory");
}

// Generic batched GEMM:  C[z] = A[z] * B[z]^T(+bias)(+resid)(relu)
//   A: [M x K] f16 row-major (lda), B: [N x K] f16 row-major (ldb, i.e. B^T).
//   z decomposed as z1 = z/zdiv, z2 = z%zdiv; operand offset = z1*Z1 + z2*Z2.
// Requires: M % 128 == 0, K % 32 == 0, 16B-aligned A/B rows (lda/ldb % 8 == 0).
template <bool NTAIL>
__global__ __launch_bounds__(256)
void wmma_gemm_async(const _Float16* __restrict__ A,
                     const _Float16* __restrict__ B,
                     float* __restrict__ Cf,
                     _Float16* __restrict__ Ch,
                     const float* __restrict__ bias,
                     const float* __restrict__ resid,
                     int M, int N, int K,
                     long long lda, long long ldb, long long ldc, long long ldch,
                     int zdiv,
                     long long aZ1, long long aZ2,
                     long long bZ1, long long bZ2,
                     long long cZ1, long long cZ2,
                     long long chZ1, long long chZ2,
                     int flags)
{
    __shared__ unsigned int ldsA[2][TM * PITCH];
    __shared__ unsigned int ldsB[2][TN * PITCH];

    const int tid   = threadIdx.x;
    const int wave  = tid >> 5;
    const int lane  = tid & 31;
    const int r     = lane & 15;
    const int khalf = lane >> 4;

    const int z = blockIdx.z;
    const long long z1 = z / zdiv;
    const long long z2 = z % zdiv;
    A += z1 * aZ1 + z2 * aZ2;
    B += z1 * bZ1 + z2 * bZ2;
    if (Cf) Cf += z1 * cZ1 + z2 * cZ2;
    if (Ch) Ch += z1 * chZ1 + z2 * chZ2;
    if (flags & FLAG_RESID) resid += z1 * cZ1 + z2 * cZ2;

    const int m0 = blockIdx.y * TM;
    const int n0 = blockIdx.x * TN;

    // ---- per-thread async staging assignment ------------------------------
    const int trow = tid >> 2;                 // 0..63
    const int tseg = (tid & 3) * 8;            // half offset in row: 0/8/16/24
    const _Float16* aRow0 = A + (long long)(m0 + trow) * lda + tseg;
    const _Float16* aRow1 = aRow0 + 64LL * lda;
    const _Float16* bRow  = B + (long long)(n0 + trow) * ldb + tseg;

    const unsigned aOffB  = (unsigned)(trow * PITCH * 4 + (tid & 3) * 16);
    const unsigned aOffB1 = aOffB + 64u * PITCH * 4u;
    const unsigned aBase0 = lds_addr(&ldsA[0][0]);
    const unsigned aBase1 = lds_addr(&ldsA[1][0]);
    const unsigned bBase0 = lds_addr(&ldsB[0][0]);
    const unsigned bBase1 = lds_addr(&ldsB[1][0]);

    v8f acc[4];
    const v8f vzero = {0.f, 0.f, 0.f, 0.f, 0.f, 0.f, 0.f, 0.f};
#pragma unroll
    for (int j = 0; j < 4; ++j) acc[j] = vzero;

    // prologue: fill buffer 0
    async_b128(aBase0 + aOffB,  aRow0);
    async_b128(aBase0 + aOffB1, aRow1);
    async_b128(bBase0 + aOffB,  bRow);

    const int nk = K / TK;
    int cur = 0;
    for (int ki = 0; ki < nk; ++ki) {
        wait_async0();
        __syncthreads();      // tile[cur] complete in LDS for all waves

        if (ki + 1 < nk) {    // prefetch next tile into the other buffer
            const int k0n = (ki + 1) * TK;
            const unsigned an = cur ? aBase0 : aBase1;
            const unsigned bn = cur ? bBase0 : bBase1;
            async_b128(an + aOffB,  aRow0 + k0n);
            async_b128(an + aOffB1, aRow1 + k0n);
            async_b128(bn + aOffB,  bRow  + k0n);
        }

        const unsigned int* sA = ldsA[cur];
        const unsigned int* sB = ldsB[cur];

        // A fragment (16-bit 16x32 layout): VGPR t holds the K pair at dword
        // col = (t&3) + 8*(t>>2) + 4*khalf of this wave's 16-row strip.
        Frag af;
#pragma unroll
        for (int t = 0; t < 8; ++t) {
            const int col = (t & 3) + ((t >> 2) << 3) + (khalf << 2);
            af.u[t] = sA[(wave * 16 + r) * PITCH + col];
        }
#pragma unroll
        for (int j = 0; j < 4; ++j) {
            Frag bf;
#pragma unroll
            for (int t = 0; t < 8; ++t) {
                const int col = (t & 3) + ((t >> 2) << 3) + (khalf << 2);
                bf.u[t] = sB[(j * 16 + r) * PITCH + col];
            }
            acc[j] = __builtin_amdgcn_wmma_f32_16x16x32_f16(
                false, af.v, false, bf.v, (short)0, acc[j], false, false);
        }
        cur ^= 1;
    }

    // ---- epilogue: C layout lane(0-15): M=v, lane(16-31): M=8+v, N=lane&15
#pragma unroll
    for (int j = 0; j < 4; ++j) {
#pragma unroll
        for (int vv = 0; vv < 8; ++vv) {
            const int m = m0 + wave * 16 + vv + (khalf << 3);
            const int n = n0 + j * 16 + r;
            if (NTAIL && n >= N) continue;
            float val = acc[j][vv];
            if (flags & FLAG_BIAS)  val += bias[n];
            if (flags & FLAG_RESID) val += resid[(long long)m * ldc + n];
            if (flags & FLAG_RELU)  val = fmaxf(val, 0.f);
            if (Cf) Cf[(long long)m * ldc + n] = val;
            if (Ch) {
                if (flags & FLAG_CHT) Ch[(long long)n * ldch + m] = (_Float16)val;
                else                  Ch[(long long)m * ldch + n] = (_Float16)val;
            }
        }
    }
}

// out[z][c*R + r0] = (f16) in[z][r0*C + c]   (convert + transpose, per batch z)
__global__ __launch_bounds__(256)
void transpose_cast_kernel(const float* __restrict__ in, _Float16* __restrict__ out,
                           int R, int C, long long inZ, long long outZ, long long total)
{
    const float* ip = in + (long long)blockIdx.z * inZ;
    _Float16* op = out + (long long)blockIdx.z * outZ;
    long long i = (long long)blockIdx.x * 256 + threadIdx.x;
    if (i >= total) return;
    long long c = i / R;
    long long r0 = i % R;
    op[c * R + r0] = (_Float16)ip[r0 * C + c];
}

// x[b,t,:] = tok_emb[idx[b,t],:] + pos_emb[t,:]   (writes f32 and f16 copies)
__global__ __launch_bounds__(256)
void embed_kernel(const int* __restrict__ idx, const float* __restrict__ tok,
                  const float* __restrict__ pos, float* __restrict__ x,
                  _Float16* __restrict__ x16, int Dn, int Tn, long long total)
{
    long long i = (long long)blockIdx.x * 256 + threadIdx.x;
    if (i >= total) return;
    long long row = i / Dn;
    int dcol = (int)(i % Dn);
    int t = (int)(row % Tn);
    float v = tok[(long long)idx[row] * Dn + dcol] + pos[(long long)t * Dn + dcol];
    x[i] = v;
    x16[i] = (_Float16)v;
}

// In-place scaled causal softmax over f16 rows of [Z*T, T] scores.
__global__ __launch_bounds__(256)
void softmax_causal_f16(_Float16* __restrict__ sc, int Tn, float scale)
{
    __shared__ float red[256];
    const long long row = blockIdx.x;
    const int t = (int)(row % Tn);
    _Float16* p = sc + row * (long long)Tn;
    const int tid = threadIdx.x;

    float mx = -INFINITY;
    for (int j = tid; j < Tn; j += 256) {
        float v = (j <= t) ? (float)p[j] * scale : -INFINITY;
        mx = fmaxf(mx, v);
    }
    red[tid] = mx; __syncthreads();
    for (int s = 128; s > 0; s >>= 1) {
        if (tid < s) red[tid] = fmaxf(red[tid], red[tid + s]);
        __syncthreads();
    }
    mx = red[0]; __syncthreads();

    float sum = 0.f;
    for (int j = tid; j < Tn; j += 256)
        sum += (j <= t) ? __expf((float)p[j] * scale - mx) : 0.f;
    red[tid] = sum; __syncthreads();
    for (int s = 128; s > 0; s >>= 1) {
        if (tid < s) red[tid] += red[tid + s];
        __syncthreads();
    }
    const float inv = 1.f / red[0];
    for (int j = tid; j < Tn; j += 256) {
        float v = (j <= t) ? __expf((float)p[j] * scale - mx) * inv : 0.f;
        p[j] = (_Float16)v;
    }
}

// rl[row] = logsumexp(logits[row,:]) - logits[row, tgt[row]]
__global__ __launch_bounds__(256)
void row_loss_kernel(const float* __restrict__ logits, const int* __restrict__ tgt,
                     float* __restrict__ rl, int Vn)
{
    __shared__ float red[256];
    const long long row = blockIdx.x;
    const float* p = logits + row * (long long)Vn;
    const int tid = threadIdx.x;

    float mx = -INFINITY;
    for (int j = tid; j < Vn; j += 256) mx = fmaxf(mx, p[j]);
    red[tid] = mx; __syncthreads();
    for (int s = 128; s > 0; s >>= 1) {
        if (tid < s) red[tid] = fmaxf(red[tid], red[tid + s]);
        __syncthreads();
    }
    mx = red[0]; __syncthreads();

    float sum = 0.f;
    for (int j = tid; j < Vn; j += 256) sum += __expf(p[j] - mx);
    red[tid] = sum; __syncthreads();
    for (int s = 128; s > 0; s >>= 1) {
        if (tid < s) red[tid] += red[tid + s];
        __syncthreads();
    }
    if (tid == 0) rl[row] = (mx + __logf(red[0])) - p[tgt[row]];
}

__global__ __launch_bounds__(256)
void loss_mean(const float* __restrict__ rl, float* __restrict__ out, int n)
{
    __shared__ float red[256];
    float s = 0.f;
    for (int j = threadIdx.x; j < n; j += 256) s += rl[j];
    red[threadIdx.x] = s; __syncthreads();
    for (int st = 128; st > 0; st >>= 1) {
        if (threadIdx.x < st) red[threadIdx.x] += red[threadIdx.x + st];
        __syncthreads();
    }
    if (threadIdx.x == 0) out[0] = red[0] / (float)n;
}

// ---------------------------------------------------------------------------

static void launch_gemm(hipStream_t stream,
                        const _Float16* A, const _Float16* B,
                        float* Cf, _Float16* Ch,
                        const float* bias, const float* resid,
                        int M, int N, int K,
                        long long lda, long long ldb, long long ldc, long long ldch,
                        int Z, int zdiv,
                        long long aZ1, long long aZ2,
                        long long bZ1, long long bZ2,
                        long long cZ1, long long cZ2,
                        long long chZ1, long long chZ2,
                        int flags)
{
    dim3 grid((N + TN - 1) / TN, (M + TM - 1) / TM, Z);
    if (N % TN == 0)
        wmma_gemm_async<false><<<grid, 256, 0, stream>>>(A, B, Cf, Ch, bias, resid,
            M, N, K, lda, ldb, ldc, ldch, zdiv,
            aZ1, aZ2, bZ1, bZ2, cZ1, cZ2, chZ1, chZ2, flags);
    else
        wmma_gemm_async<true><<<grid, 256, 0, stream>>>(A, B, Cf, Ch, bias, resid,
            M, N, K, lda, ldb, ldc, ldch, zdiv,
            aZ1, aZ2, bZ1, bZ2, cZ1, cZ2, chZ1, chZ2, flags);
}

static void launch_transpose(hipStream_t stream, const float* in, _Float16* out,
                             int R, int C, long long inZ, long long outZ, int Z)
{
    const long long total = (long long)R * C;
    dim3 grid((unsigned)((total + 255) / 256), 1, Z);
    transpose_cast_kernel<<<grid, 256, 0, stream>>>(in, out, R, C, inZ, outZ, total);
}

extern "C" void kernel_launch(void* const* d_in, const int* in_sizes, int n_in,
                              void* d_out, int out_size, void* d_ws, size_t ws_size,
                              hipStream_t stream)
{
    (void)in_sizes; (void)n_in; (void)out_size; (void)ws_size;

    const int Vn = 50257, Dn = 768, Hn = 12, HSn = 64, Tn = 1024, Ln = 12, Bn = 4;
    const int Mr = Bn * Tn;                       // 4096 token rows
    const long long D4 = 4LL * Dn;                // 3072

    const int*   idx     = (const int*)d_in[0];
    const int*   targets = (const int*)d_in[1];
    const float* tok     = (const float*)d_in[2];
    const float* pos     = (const float*)d_in[3];
    const float* Wq      = (const float*)d_in[4];
    const float* Wk      = (const float*)d_in[5];
    const float* Wv      = (const float*)d_in[6];
    const float* Wo      = (const float*)d_in[7];
    const float* bo      = (const float*)d_in[8];
    const float* W1      = (const float*)d_in[9];
    const float* b1      = (const float*)d_in[10];
    const float* W2      = (const float*)d_in[11];
    const float* b2      = (const float*)d_in[12];
    const float* lmW     = (const float*)d_in[13];
    const float* lmb     = (const float*)d_in[14];

    float* out = (float*)d_out;                   // [Mr*Vn logits] + [1 loss]

    // -------- workspace layout ------------------------------------------
    float* x     = (float*)d_ws;                               // Mr*Dn f32
    float* rloss = x + (long long)Mr * Dn;                     // Mr f32
    _Float16* h0 = (_Float16*)(rloss + Mr);

    _Float16* x16   = h0;                                      // Mr*Dn
    _Float16* af16  = x16  + (long long)Mr * Dn;               // Mr*Dn
    _Float16* h16   = af16 + (long long)Mr * Dn;               // Mr*4Dn
    _Float16* q16   = h16  + (long long)Mr * D4;               // Hn*Mr*HSn
    _Float16* k16   = q16  + (long long)Hn * Mr * HSn;
    _Float16* vT16  = k16  + (long long)Hn * Mr * HSn;         // [Hn][HSn][Mr]
    _Float16* sc16  = vT16 + (long long)Hn * Mr * HSn;         // Hn*Bn*Tn*Tn
    _Float16* wqT   = sc16 + (long long)Hn * Bn * Tn * Tn;     // Ln*Hn*HSn*Dn
    _Float16* wkT   = wqT  + (long long)Ln * Hn * HSn * Dn;
    _Float16* wvT   = wkT  + (long long)Ln * Hn * HSn * Dn;
    _Float16* woT   = wvT  + (long long)Ln * Hn * HSn * Dn;    // Ln*Dn*Dn
    _Float16* w1T   = woT  + (long long)Ln * Dn * Dn;          // Ln*4Dn*Dn
    _Float16* w2T   = w1T  + (long long)Ln * D4 * Dn;          // Ln*Dn*4Dn
    _Float16* lmWT  = w2T  + (long long)Ln * Dn * D4;          // Vn*Dn (+slack after)

    // -------- 0. weights -> f16, transposed to [N x K] -------------------
    launch_transpose(stream, Wq, wqT, Dn, HSn, (long long)Dn * HSn, (long long)Dn * HSn, Ln * Hn);
    launch_transpose(stream, Wk, wkT, Dn, HSn, (long long)Dn * HSn, (long long)Dn * HSn, Ln * Hn);
    launch_transpose(stream, Wv, wvT, Dn, HSn, (long long)Dn * HSn, (long long)Dn * HSn, Ln * Hn);
    launch_transpose(stream, Wo, woT, Dn, Dn, (long long)Dn * Dn, (long long)Dn * Dn, Ln);
    launch_transpose(stream, W1, w1T, Dn, (int)D4, (long long)Dn * D4, (long long)Dn * D4, Ln);
    launch_transpose(stream, W2, w2T, (int)D4, Dn, (long long)Dn * D4, (long long)Dn * D4, Ln);
    launch_transpose(stream, lmW, lmWT, Dn, Vn, 0, 0, 1);

    // -------- 1. embedding ----------------------------------------------
    {
        const long long total = (long long)Mr * Dn;
        embed_kernel<<<(unsigned)((total + 255) / 256), 256, 0, stream>>>(
            idx, tok, pos, x, x16, Dn, Tn, total);
    }

    const float scale = 1.0f / sqrtf((float)Dn);  // reference scales by D^-0.5

    for (int l = 0; l < Ln; ++l) {
        const long long wOff = (long long)l * Hn * HSn * Dn;

        // 2. Q,K projections (Z = H); V stored transposed [h][e][r]
        launch_gemm(stream, x16, wqT + wOff, nullptr, q16, nullptr, nullptr,
                    Mr, HSn, Dn, Dn, Dn, 0, HSn,
                    Hn, 1, 0, 0, (long long)HSn * Dn, 0, 0, 0,
                    (long long)Mr * HSn, 0, 0);
        launch_gemm(stream, x16, wkT + wOff, nullptr, k16, nullptr, nullptr,
                    Mr, HSn, Dn, Dn, Dn, 0, HSn,
                    Hn, 1, 0, 0, (long long)HSn * Dn, 0, 0, 0,
                    (long long)Mr * HSn, 0, 0);
        launch_gemm(stream, x16, wvT + wOff, nullptr, vT16, nullptr, nullptr,
                    Mr, HSn, Dn, Dn, Dn, 0, (long long)Mr,
                    Hn, 1, 0, 0, (long long)HSn * Dn, 0, 0, 0,
                    (long long)HSn * Mr, 0, FLAG_CHT);

        // 3. scores = Q @ K^T per (h,b): B = k16 is already [N=s][K=e]
        launch_gemm(stream, q16, k16, nullptr, sc16, nullptr, nullptr,
                    Tn, Tn, HSn, HSn, HSn, 0, Tn,
                    Hn * Bn, 1,
                    (long long)Tn * HSn, 0,
                    (long long)Tn * HSn, 0, 0, 0,
                    (long long)Tn * Tn, 0, 0);

        // 4. causal softmax (scale fused), in place on f16 scores
        softmax_causal_f16<<<(unsigned)(Hn * Bn * Tn), 256, 0, stream>>>(sc16, Tn, scale);

        // 5. attn = P @ V: B = vT16 [e][r] rows are K(=s)-contiguous
        launch_gemm(stream, sc16, vT16, nullptr, af16, nullptr, nullptr,
                    Tn, HSn, Tn, Tn, Mr, 0, (long long)Hn * HSn,
                    Hn * Bn, Bn,
                    (long long)Bn * Tn * Tn, (long long)Tn * Tn,
                    (long long)HSn * Mr,     (long long)Tn,
                    0, 0,
                    (long long)HSn,          (long long)Tn * Hn * HSn,
                    0);

        // 6. x = x + attn @ Wo + bo   (f32 residual stream + f16 copy)
        launch_gemm(stream, af16, woT + (long long)l * Dn * Dn, x, x16,
                    bo + (long long)l * Dn, x,
                    Mr, Dn, Dn, (long long)Hn * HSn, Dn, Dn, Dn,
                    1, 1, 0, 0, 0, 0, 0, 0, 0, 0, FLAG_BIAS | FLAG_RESID);

        // 7. h = relu(x @ W1 + b1) -> f16
        launch_gemm(stream, x16, w1T + (long long)l * D4 * Dn, nullptr, h16,
                    b1 + (long long)l * D4, nullptr,
                    Mr, (int)D4, Dn, Dn, Dn, 0, D4,
                    1, 1, 0, 0, 0, 0, 0, 0, 0, 0, FLAG_BIAS | FLAG_RELU);

        // 8. x = x + h @ W2 + b2
        launch_gemm(stream, h16, w2T + (long long)l * Dn * D4, x, x16,
                    b2 + (long long)l * Dn, x,
                    Mr, Dn, (int)D4, D4, D4, Dn, Dn,
                    1, 1, 0, 0, 0, 0, 0, 0, 0, 0, FLAG_BIAS | FLAG_RESID);
    }

    // 9. logits = x @ lm_W + lm_b -> d_out (N-tail path)
    launch_gemm(stream, x16, lmWT, out, nullptr, lmb, nullptr,
                Mr, Vn, Dn, Dn, Dn, Vn, 0,
                1, 1, 0, 0, 0, 0, 0, 0, 0, 0, FLAG_BIAS);

    // 10. loss = mean(logsumexp - logit[target])
    row_loss_kernel<<<Mr, 256, 0, stream>>>(out, targets, rloss, Vn);
    loss_mean<<<1, 256, 0, stream>>>(rloss, out + (long long)Mr * Vn, Mr);
}
